// Decoder_60224031425226
// MI455X (gfx1250) — matrix operand order repaired and checked by reading, not compiled, not run on hardware
//
#include <hip/hip_runtime.h>
#include <hip/hip_bf16.h>

// ---------------------------------------------------------------------------
// MI455X (gfx1250, wave32) implementation.
// - All dense GEMMs: v_wmma_f32_16x16x32_f16 (f16 in, f32 accumulate).
// - B (weight) panel of each GEMM block is staged into LDS by the Tensor Data
//   Mover (TENSOR_LOAD_TO_LDS + s_wait_tensorcnt), shared by all 8 waves.
// - A/B WMMA fragments use an explicit 2-deep pipeline with NAMED variables
//   (no register arrays -> no v_movrels) and a manual 2x K unroll.
// - LSTM recurrence: one persistent 32-wave workgroup, h in LDS as f16,
//   WMMA per step, gates on VALU.
// - Attention fused (tanh/softmax/context) on VALU+TRANS, no 536MB tensor.
// - Final vocab GEMM is HBM-store-bound (~165MB -> ~7us @ 23.3TB/s), so f16
//   WMMA is already past the memory roofline.
// ---------------------------------------------------------------------------

typedef _Float16 h16;
typedef __attribute__((ext_vector_type(16))) _Float16 v16h;
typedef __attribute__((ext_vector_type(8)))  _Float16 v8h;
typedef __attribute__((ext_vector_type(8)))  float    v8f;

constexpr int kTq = 128, kTk = 256, kB = 8, kD = 512, kH4 = 2048, kV = 32000;

__device__ __forceinline__ float sigm(float x) { return 1.0f / (1.0f + __expf(-x)); }

#define LOAD_A(frag, base, off)                                   \
  do {                                                            \
    ((v8h*)&(frag))[0] = *(const v8h*)((base) + (off));           \
    ((v8h*)&(frag))[1] = *(const v8h*)((base) + (off) + 16);      \
  } while (0)
#define LOAD_B(frag, base, off)                                   \
  do {                                                            \
    ((v8h*)&(frag))[0] = *(const v8h*)((base) + (off));           \
    ((v8h*)&(frag))[1] = *(const v8h*)((base) + (off) + 16);      \
  } while (0)

// ---------------- elementwise helpers ----------------
__global__ __launch_bounds__(256)
void cvt_f32_h16(const float* __restrict__ s, h16* __restrict__ d, int n) {
  int i = blockIdx.x * 256 + threadIdx.x;
  if (i < n) d[i] = (h16)s[i];
}

__global__ __launch_bounds__(256)
void embed_k(const int* __restrict__ tok, const float* __restrict__ emb,
             float* __restrict__ t, int n) {
  int i = blockIdx.x * 256 + threadIdx.x;
  if (i < n) {
    int row = i >> 9, d = i & 511;                  // (Tq*B, D)
    t[i] = emb[(size_t)tok[row] * kD + d];
  }
}

__global__ __launch_bounds__(256)
void concat_k(const float* __restrict__ t, const float* __restrict__ ctx,
              h16* __restrict__ x, int n) {
  int i = blockIdx.x * 256 + threadIdx.x;
  if (i < n) {
    int row = i >> 10, c = i & 1023;                // (Tq*B, 2D)
    float v = (c < kD) ? t[(size_t)row * kD + c] : ctx[(size_t)row * kD + (c - kD)];
    x[i] = (h16)v;
  }
}

// ---------------- generic WMMA GEMM: C = A(MxK) * B(NxK)^T + bias ----------
// A,B row-major f16 (K contiguous -> matches WMMA A/B lane layouts exactly).
// block = 256 threads = 8 waves; wave -> one 16x16 tile; block covers 128 rows.
// The 16-row B panel (shared by all 8 waves) is DMA'd to LDS by the TDM.
// Requires K % 64 == 0 (all call sites: 512 or 1024).
__global__ __launch_bounds__(256)
void gemm_wmma(const h16* __restrict__ A, const h16* __restrict__ B,
               const float* __restrict__ bias1, const float* __restrict__ bias2,
               float* __restrict__ C, int M, int N, int K) {
  __shared__ h16 bS[16 * 1024];                     // 16 rows x (K<=1024) halves
  const int nBase = blockIdx.x * 16;

#if __has_builtin(__builtin_amdgcn_tensor_load_to_lds)
  // --- TDM: stage B[nBase..nBase+16) x K into LDS (one DMA per block) ---
  if (threadIdx.x < 32) {                           // TDM ignores EXEC; issue once
    typedef __attribute__((ext_vector_type(4))) unsigned int u32x4;
    typedef __attribute__((ext_vector_type(8))) int          i32x8;
    typedef __attribute__((ext_vector_type(4))) int          i32x4;
    unsigned long long ga = (unsigned long long)(uintptr_t)(B + (size_t)nBase * K);
    unsigned lds = (unsigned)(uintptr_t)(void*)bS;  // addr[31:0] = LDS offset

    u32x4 g0;
    g0[0] = 1u;                                     // count=1, user descriptor
    g0[1] = lds;                                    // lds_addr   (bits 63:32)
    g0[2] = (unsigned)(ga & 0xffffffffu);           // global_addr[31:0]
    g0[3] = (unsigned)((ga >> 32) & 0x01ffffffu)    // global_addr[56:32]
          | (2u << 30);                             // type=2 ("image")

    i32x8 g1;
    g1[0] = (int)(1u << 16);                        // wg_mask=0, data_size=1 (2B)
    g1[1] = (int)(((unsigned)K & 0xffffu) << 16);   // tensor_dim0[15:0] @ 63:48
    g1[2] = (int)((((unsigned)K >> 16) & 0xffffu)   // tensor_dim0[31:16]
          | (((unsigned)N & 0xffffu) << 16));       // tensor_dim1[15:0]
    g1[3] = (int)((((unsigned)N >> 16) & 0xffffu)   // tensor_dim1[31:16]
          | (((unsigned)K & 0xffffu) << 16));       // tile_dim0 = K
    g1[4] = 16;                                     // tile_dim1=16, tile_dim2=0
    g1[5] = K;                                      // tensor_dim0_stride[31:0]
    g1[6] = 0;                                      // stride hi / dim1_stride lo
    g1[7] = 0;

    i32x4 z4 = {0, 0, 0, 0};                        // groups 2/3 unused (2-D)
    i32x8 z8 = {0, 0, 0, 0, 0, 0, 0, 0};            // extra group (clang-23 form)
    __builtin_amdgcn_tensor_load_to_lds(g0, g1, z4, z4, z8, 0);
    __builtin_amdgcn_s_wait_tensorcnt(0);
  }
#else
  for (int i = threadIdx.x; i < 16 * K; i += 256)
    bS[i] = B[(size_t)nBase * K + i];
#endif
  __syncthreads();

  const int wave = threadIdx.x >> 5;
  const int lane = threadIdx.x & 31;
  const int lh = lane & 15, lg = lane >> 4;
  const int mBase = blockIdx.y * 128 + wave * 16;
  if (mBase >= M) return;

  // A: lane = row (m), lane-group selects K sub-bank (ISA 7.12.2 16-bit A layout)
  const h16* aRow = A + (size_t)(mBase + lh) * K + lg * 8;
  // B: lane = column (n), from LDS panel; lane-group selects K sub-bank (same as A)
  const h16* bRow = bS + (size_t)lh * K + lg * 8;

  // explicit 2-deep pipeline, named fragments (no register arrays)
  v16h a0, b0, a1, b1;
  LOAD_A(a0, aRow, 0);
  LOAD_B(b0, bRow, 0);

  v8f acc = {};
  for (int k = 0; k < K; k += 64) {
    LOAD_A(a1, aRow, k + 32);
    LOAD_B(b1, bRow, k + 32);
    __builtin_prefetch(aRow + k + 256, 0, 1);       // global_prefetch_b8
    acc = __builtin_amdgcn_wmma_f32_16x16x32_f16(false, a0, false, b0,
                                                 (short)0, acc, false, false);
    const int kn = (k + 64 < K) ? (k + 64) : 0;     // tail: dummy reload @0
    LOAD_A(a0, aRow, kn);
    LOAD_B(b0, bRow, kn);
    acc = __builtin_amdgcn_wmma_f32_16x16x32_f16(false, a1, false, b1,
                                                 (short)0, acc, false, false);
  }

  // C/D layout: lanes 0-15 -> rows m..m+7, lanes 16-31 -> rows m+8..m+15
  const int col = nBase + lh;
  float bv = (bias1 ? bias1[col] : 0.0f) + (bias2 ? bias2[col] : 0.0f);
  float* cp = C + (size_t)(mBase + lg * 8) * N + col;
#pragma unroll
  for (int r = 0; r < 8; ++r) cp[(size_t)r * N] = acc[r] + bv;
}

// ---------------- persistent LSTM recurrence ----------------
// xproj = x@Wih^T + bih + bhh precomputed for all steps. This kernel runs the
// sequential part: g = xproj[step] + h @ Whh^T ; gates ; h,c update.
// h lives in LDS as f16 (WMMA A operand, rows 8..15 zero-padded).
__global__ __launch_bounds__(1024)
void lstm_rec(const float* __restrict__ xproj,   // (T*8, 2048)
              const float* __restrict__ h0,      // (8, 512)
              const float* __restrict__ c0,      // (8, 512)
              const h16*   __restrict__ Whh,     // (2048, 512) f16
              const float* __restrict__ addIn,   // residual input (may alias out), or null
              float* __restrict__ out,           // (T*8, 512)
              int T) {
  __shared__ h16   hS[8][520];                    // +8 halves pad, rows stay 16B aligned
  __shared__ float cS[8][512];
  __shared__ float gS[8][2052];

  const int tid = threadIdx.x;
  for (int i = tid; i < 8 * kD; i += 1024) {
    int b = i >> 9, d = i & 511;
    hS[b][d] = (h16)h0[i];
    cS[b][d] = c0[i];
  }
  __syncthreads();

  const int wave = tid >> 5, lane = tid & 31;
  const int lh = lane & 15, lg = lane >> 4;

  for (int step = 0; step < T; ++step) {
    // ---- g[0:8, 0:2048] = h @ Whh^T  (WMMA, M padded to 16) ----
    for (int t4 = 0; t4 < 4; ++t4) {
      const int nBase = (wave * 4 + t4) * 16;
      const h16* bRow = Whh + (size_t)(nBase + lh) * kD + lg * 8;
      const h16* aLds = &hS[lh][lg * 8];

      v16h a0 = {}, a1 = {}, b0, b1;                // zero A rows 8..15 persist
      if (lh < 8) LOAD_A(a0, aLds, 0);
      LOAD_B(b0, bRow, 0);

      v8f acc = {};
#pragma unroll
      for (int k = 0; k < kD; k += 64) {
        if (lh < 8) LOAD_A(a1, aLds, k + 32);
        LOAD_B(b1, bRow, k + 32);
        acc = __builtin_amdgcn_wmma_f32_16x16x32_f16(false, a0, false, b0,
                                                     (short)0, acc, false, false);
        const int kn = (k + 64 < kD) ? (k + 64) : 0;
        if (lh < 8) LOAD_A(a0, aLds, kn);
        LOAD_B(b0, bRow, kn);
        acc = __builtin_amdgcn_wmma_f32_16x16x32_f16(false, a1, false, b1,
                                                     (short)0, acc, false, false);
      }
      if (lg == 0) {                                // lanes 0-15 hold valid rows 0..7
        const int col = nBase + lh;
        const float* xp = xproj + (size_t)step * 8 * kH4 + col;
#pragma unroll
        for (int r = 0; r < 8; ++r) gS[r][col] = acc[r] + xp[(size_t)r * kH4];
      }
    }
    __syncthreads();

    // ---- gates (PyTorch order i,f,g,o) ----
    for (int i = tid; i < 8 * kD; i += 1024) {
      int b = i >> 9, d = i & 511;
      float gi = gS[b][d];
      float gf = gS[b][512 + d];
      float gg = gS[b][1024 + d];
      float go = gS[b][1536 + d];
      float c = sigm(gf) * cS[b][d] + sigm(gi) * tanhf(gg);
      float h = sigm(go) * tanhf(c);
      cS[b][d] = c;
      hS[b][d] = (h16)h;
      size_t o = (size_t)(step * 8 + b) * kD + d;
      out[o] = addIn ? (addIn[o] + h) : h;
    }
    __syncthreads();
  }
}

// ---------------- fused additive attention ----------------
// one block per (q,b): scores (tanh+dot), mask, softmax over Tk, context.
__global__ __launch_bounds__(256)
void attn_k(const float* __restrict__ qp, const float* __restrict__ kp,
            const float* __restrict__ vp, const unsigned char* __restrict__ mask,
            const float* __restrict__ Ww, const float* __restrict__ Wb,
            float* __restrict__ ctx, int accumulate) {
  __shared__ float qS[512], WwS[512], wS[256], red[256];
  const int q = blockIdx.x, b = blockIdx.y, tid = threadIdx.x;

  for (int a = tid; a < kD; a += 256) {
    qS[a]  = qp[((size_t)q * kB + b) * kD + a];
    WwS[a] = Ww[a];
  }
  __syncthreads();

  // score for k = tid
  const float* kr = kp + ((size_t)tid * kB + b) * kD;
  float s = 0.0f;
  for (int a = 0; a < kD; ++a) s += tanhf(qS[a] + kr[a]) * WwS[a];
  s += Wb[0];
  if (!mask[((size_t)q * kTk + tid) * kB + b]) s = -1.0e6f;

  // softmax over Tk
  red[tid] = s; __syncthreads();
  for (int off = 128; off > 0; off >>= 1) {
    if (tid < off) red[tid] = fmaxf(red[tid], red[tid + off]);
    __syncthreads();
  }
  float mx = red[0]; __syncthreads();
  float e = __expf(s - mx);
  red[tid] = e; __syncthreads();
  for (int off = 128; off > 0; off >>= 1) {
    if (tid < off) red[tid] += red[tid + off];
    __syncthreads();
  }
  float inv = 1.0f / red[0];
  wS[tid] = e * inv;
  __syncthreads();

  // ctx[q,b,a] = sum_k w[k] * vp[k,b,a]
  for (int a = tid; a < kD; a += 256) {
    float acc = 0.0f;
    for (int k = 0; k < kTk; ++k) acc += wS[k] * vp[((size_t)k * kB + b) * kD + a];
    float* dst = ctx + ((size_t)q * kB + b) * kD + a;
    if (accumulate) *dst += acc; else *dst = acc;
  }
}

// ---------------------------------------------------------------------------
extern "C" void kernel_launch(void* const* d_in, const int* in_sizes, int n_in,
                              void* d_out, int out_size, void* d_ws, size_t ws_size,
                              hipStream_t stream) {
  (void)in_sizes; (void)n_in; (void)out_size; (void)ws_size;

  const int*   inputs  = (const int*)  d_in[0];
  const float* h0      = (const float*)d_in[1];
  const float* c0      = (const float*)d_in[2];
  const float* enc1    = (const float*)d_in[3];
  const float* enc2    = (const float*)d_in[4];
  const unsigned char* mask1 = (const unsigned char*)d_in[5];
  const unsigned char* mask2 = (const unsigned char*)d_in[6];
  const float* emb     = (const float*)d_in[7];
  const float* Wih_res = (const float*)d_in[8];
  const float* Whh_res = (const float*)d_in[9];
  const float* bih_res = (const float*)d_in[10];
  const float* bhh_res = (const float*)d_in[11];
  const float* WihF    = (const float*)d_in[12];
  const float* WhhF    = (const float*)d_in[13];
  const float* bihF    = (const float*)d_in[14];
  const float* bhhF    = (const float*)d_in[15];
  const float* fcw     = (const float*)d_in[16];
  const float* fcb     = (const float*)d_in[17];
  const float* Qw1 = (const float*)d_in[18], *Qb1 = (const float*)d_in[19];
  const float* Kw1 = (const float*)d_in[20], *Kb1 = (const float*)d_in[21];
  const float* Vw1 = (const float*)d_in[22], *Vb1 = (const float*)d_in[23];
  const float* Ww1 = (const float*)d_in[24], *Wb1 = (const float*)d_in[25];
  const float* Qw2 = (const float*)d_in[26], *Qb2 = (const float*)d_in[27];
  const float* Kw2 = (const float*)d_in[28], *Kb2 = (const float*)d_in[29];
  const float* Vw2 = (const float*)d_in[30], *Vb2 = (const float*)d_in[31];
  const float* Ww2 = (const float*)d_in[32], *Wb2 = (const float*)d_in[33];

  // ---- workspace bump allocator (~96 MB total) ----
  size_t used = 0;
  auto alloc = [&](size_t bytes) -> void* {
    void* p = (char*)d_ws + used;
    used += (bytes + 255) & ~(size_t)255;
    return p;
  };
  const int MB = kTq * kB;          // 1024 rows for decoder-side matrices
  const int ME = kTk * kB;          // 2048 rows for encoder-side matrices

  h16* wWih0 = (h16*)alloc((size_t)kH4 * kD * 2);
  h16* wWih1 = (h16*)alloc((size_t)kH4 * kD * 2);
  h16* wWhh0 = (h16*)alloc((size_t)kH4 * kD * 2);
  h16* wWhh1 = (h16*)alloc((size_t)kH4 * kD * 2);
  h16* wWihF = (h16*)alloc((size_t)kH4 * 2 * kD * 2);
  h16* wWhhF = (h16*)alloc((size_t)kH4 * kD * 2);
  h16* hQw1  = (h16*)alloc((size_t)kD * kD * 2);
  h16* hKw1  = (h16*)alloc((size_t)kD * kD * 2);
  h16* hVw1  = (h16*)alloc((size_t)kD * kD * 2);
  h16* hQw2  = (h16*)alloc((size_t)kD * kD * 2);
  h16* hKw2  = (h16*)alloc((size_t)kD * kD * 2);
  h16* hVw2  = (h16*)alloc((size_t)kD * kD * 2);
  h16* hFcw  = (h16*)alloc((size_t)kV * kD * 2);
  h16* hEnc1 = (h16*)alloc((size_t)ME * kD * 2);
  h16* hEnc2 = (h16*)alloc((size_t)ME * kD * 2);
  h16* t_h   = (h16*)alloc((size_t)MB * kD * 2);
  h16* x_h   = (h16*)alloc((size_t)MB * 2 * kD * 2);
  h16* hF_h  = (h16*)alloc((size_t)MB * kD * 2);

  float* tbuf  = (float*)alloc((size_t)MB * kD * 4);
  float* xproj = (float*)alloc((size_t)MB * kH4 * 4);
  float* qp1   = (float*)alloc((size_t)MB * kD * 4);
  float* qp2   = (float*)alloc((size_t)MB * kD * 4);
  float* kp1   = (float*)alloc((size_t)ME * kD * 4);
  float* vp1   = (float*)alloc((size_t)ME * kD * 4);
  float* kp2   = (float*)alloc((size_t)ME * kD * 4);
  float* vp2   = (float*)alloc((size_t)ME * kD * 4);
  float* ctx   = (float*)alloc((size_t)MB * kD * 4);
  float* hF    = (float*)alloc((size_t)MB * kD * 4);

  auto cvt = [&](const float* s, h16* d, int n) {
    cvt_f32_h16<<<(n + 255) / 256, 256, 0, stream>>>(s, d, n);
  };
  auto gemm = [&](const h16* A, const h16* Bm, const float* b1, const float* b2,
                  float* C, int M, int N, int K) {
    dim3 g(N / 16, M / 128);
    gemm_wmma<<<g, 256, 0, stream>>>(A, Bm, b1, b2, C, M, N, K);
  };

  // ---- weight / input conversions to f16 ----
  cvt(Wih_res,             wWih0, kH4 * kD);
  cvt(Wih_res + kH4 * kD,  wWih1, kH4 * kD);
  cvt(Whh_res,             wWhh0, kH4 * kD);
  cvt(Whh_res + kH4 * kD,  wWhh1, kH4 * kD);
  cvt(WihF,  wWihF, kH4 * 2 * kD);
  cvt(WhhF,  wWhhF, kH4 * kD);
  cvt(Qw1, hQw1, kD * kD);  cvt(Kw1, hKw1, kD * kD);  cvt(Vw1, hVw1, kD * kD);
  cvt(Qw2, hQw2, kD * kD);  cvt(Kw2, hKw2, kD * kD);  cvt(Vw2, hVw2, kD * kD);
  cvt(fcw,  hFcw,  kV * kD);
  cvt(enc1, hEnc1, ME * kD);
  cvt(enc2, hEnc2, ME * kD);

  // ---- embedding ----
  embed_k<<<(MB * kD + 255) / 256, 256, 0, stream>>>(inputs, emb, tbuf, MB * kD);

  // ---- residual LSTM layer 0 ----
  cvt(tbuf, t_h, MB * kD);
  gemm(t_h, wWih0, bih_res, bhh_res, xproj, MB, kH4, kD);
  lstm_rec<<<1, 1024, 0, stream>>>(xproj, h0, c0, wWhh0, tbuf, tbuf, kTq);

  // ---- residual LSTM layer 1 ----
  cvt(tbuf, t_h, MB * kD);
  gemm(t_h, wWih1, bih_res + kH4, bhh_res + kH4, xproj, MB, kH4, kD);
  lstm_rec<<<1, 1024, 0, stream>>>(xproj, h0 + kB * kD, c0 + kB * kD, wWhh1,
                                   tbuf, tbuf, kTq);

  // ---- attention projections (all WMMA) ----
  cvt(tbuf, t_h, MB * kD);
  gemm(t_h,   hQw1, Qb1, nullptr, qp1, MB, kD, kD);
  gemm(t_h,   hQw2, Qb2, nullptr, qp2, MB, kD, kD);
  gemm(hEnc1, hKw1, Kb1, nullptr, kp1, ME, kD, kD);
  gemm(hEnc1, hVw1, Vb1, nullptr, vp1, ME, kD, kD);
  gemm(hEnc2, hKw2, Kb2, nullptr, kp2, ME, kD, kD);
  gemm(hEnc2, hVw2, Vb2, nullptr, vp2, ME, kD, kD);

  // ---- fused attention (scores+mask+softmax+context); ctx = c1 + c2 ----
  dim3 ag(kTq, kB);
  attn_k<<<ag, 256, 0, stream>>>(qp1, kp1, vp1, mask1, Ww1, Wb1, ctx, 0);
  attn_k<<<ag, 256, 0, stream>>>(qp2, kp2, vp2, mask2, Ww2, Wb2, ctx, 1);

  // ---- concat [t, c1+c2] -> f16 ----
  concat_k<<<(MB * 2 * kD + 255) / 256, 256, 0, stream>>>(tbuf, ctx, x_h, MB * 2 * kD);

  // ---- final LSTM ----
  gemm(x_h, wWihF, bihF, bhhF, xproj, MB, kH4, 2 * kD);
  lstm_rec<<<1, 1024, 0, stream>>>(xproj, h0 + 2 * kB * kD, c0 + 2 * kB * kD,
                                   wWhhF, nullptr, hF, kTq);

  // ---- vocab projection (HBM-bound: ~165MB traffic) ----
  cvt(hF, hF_h, MB * kD);
  gemm(hF_h, hFcw, fcb, nullptr, (float*)d_out, MB, kV, kD);
}
